// RecurrentBinaryQuantizer_31482110279874
// MI455X (gfx1250) — matrix operand order repaired and checked by reading, not compiled, not run on hardware
//
#include <hip/hip_runtime.h>

typedef __attribute__((ext_vector_type(16))) __bf16 v16bf;
typedef __attribute__((ext_vector_type(8)))  float  v8f;

union FragBF {
    uint4  u[2];   // 32 bytes
    v16bf  v;
};

__device__ __forceinline__ unsigned short f2bf(float f) {
    // round-to-nearest-even f32 -> bf16
    unsigned u = __float_as_uint(f);
    u = u + 0x7FFFu + ((u >> 16) & 1u);
    return (unsigned short)(u >> 16);
}

#define Hdim 256
#define NG   768          // 3*H
#define NOUT 12
#define BLrows 32768      // B*L
#define ROWS_PER_WAVE 16
#define WAVES_PER_WG  4
#define ROWS_PER_WG   64  // 4 waves * 16 rows

// -------- pre-pass: convert w_hh [768,256] f32 -> bf16 in workspace --------
__global__ void convert_whh_bf16(const float* __restrict__ w,
                                 unsigned short* __restrict__ out, int n) {
    int i = blockIdx.x * blockDim.x + threadIdx.x;
    if (i < n) out[i] = f2bf(w[i]);
}

// -------- main persistent GRU kernel: one wave = 16 rows, 12 steps --------
// launch_bounds(128, 1): 4 waves -> 1 wave/SIMD, full VGPR budget, no spills
__global__ void __launch_bounds__(128, 1)
gru_wmma_kernel(const float* __restrict__ E,      // [BL,256]
                const float* __restrict__ w_ih,   // [768]
                const float* __restrict__ b_ih,   // [768]
                const float* __restrict__ b_hh,   // [768]
                const float* __restrict__ w_out,  // [256]
                const float* __restrict__ b_out,  // [1]
                const unsigned short* __restrict__ Wbf, // [768,256] bf16
                float* __restrict__ out)          // [BL*12] out | [BL*12] quant
{
    extern __shared__ char smem[];
    const int tid  = threadIdx.x;
    const int lane = tid & 31;
    const int wave = tid >> 5;
    const int hi   = (lane >> 4) & 1;   // lane half (WMMA layout selector)
    const int ln16 = lane & 15;

    float*          h_all = (float*)smem;                         // 64*256 f32
    float*          e_all = h_all + ROWS_PER_WG * Hdim;           // 64*256 f32
    unsigned short* a_all = (unsigned short*)(e_all + ROWS_PER_WG * Hdim); // 64*256 bf16

    float*          h_w = h_all + wave * (ROWS_PER_WAVE * Hdim);
    float*          e_w = e_all + wave * (ROWS_PER_WAVE * Hdim);
    unsigned short* a_w = a_all + wave * (ROWS_PER_WAVE * Hdim);

    const int row0 = blockIdx.x * ROWS_PER_WG + wave * ROWS_PER_WAVE;

    // load embedding tile (16 rows = 4096 contiguous floats), zero h
    const float4* esrc = (const float4*)(E + (size_t)row0 * Hdim);
    #pragma unroll
    for (int t = 0; t < 32; ++t) {
        int i = t * 32 + lane;
        float4 ev = esrc[i];
        *(float4*)&e_w[i * 4] = ev;
        float4 z4; z4.x = z4.y = z4.z = z4.w = 0.0f;
        *(float4*)&h_w[i * 4] = z4;
    }

    const float bout = b_out[0];
    const int   BLN  = BLrows * NOUT;

    // quantized feedback for "my" 8 rows (row = v + 8*hi); x0 = 0
    float xq[8];
    #pragma unroll
    for (int v = 0; v < 8; ++v) xq[v] = 0.0f;

    for (int s = 0; s < NOUT; ++s) {
        // ---- phase 1: stage A = bf16(h + e) into LDS (row-major 16x256) ----
        #pragma unroll
        for (int t = 0; t < 32; ++t) {
            int i = (t * 32 + lane) * 4;
            float4 hv = *(const float4*)&h_w[i];
            float4 ev = *(const float4*)&e_w[i];
            uint2 p;
            p.x = (unsigned)f2bf(hv.x + ev.x) | ((unsigned)f2bf(hv.y + ev.y) << 16);
            p.y = (unsigned)f2bf(hv.z + ev.z) | ((unsigned)f2bf(hv.w + ev.w) << 16);
            *(uint2*)&a_w[i] = p;
        }

        // ---- load 8 resident A fragments (K = 0..255), contiguous k = 16*hi + s layout ----
        FragBF afr[8];
        #pragma unroll
        for (int kc = 0; kc < 8; ++kc) {
            int base = ln16 * Hdim + kc * 32 + hi * 16;
            afr[kc].u[0] = *(const uint4*)&a_w[base];
            afr[kc].u[1] = *(const uint4*)&a_w[base + 8];
        }

        float op[8];
        #pragma unroll
        for (int v = 0; v < 8; ++v) op[v] = 0.0f;

        // ---- 16 column tiles of 16 over H; 3 gate accumulators each ----
        for (int j = 0; j < 16; ++j) {
            const int c = j * 16 + ln16;        // hidden column this lane owns in C
            v8f accR = {}; v8f accZ = {}; v8f accN = {};
            #pragma unroll
            for (int kc = 0; kc < 8; ++kc) {
                const size_t koff = (size_t)kc * 32 + (size_t)hi * 16;
                const unsigned short* wr = Wbf + ((size_t)(c)       * Hdim) + koff;
                const unsigned short* wz = Wbf + ((size_t)(256 + c) * Hdim) + koff;
                const unsigned short* wn = Wbf + ((size_t)(512 + c) * Hdim) + koff;
                FragBF br, bz, bn;
                br.u[0] = *(const uint4*)wr;  br.u[1] = *(const uint4*)(wr + 8);
                bz.u[0] = *(const uint4*)wz;  bz.u[1] = *(const uint4*)(wz + 8);
                bn.u[0] = *(const uint4*)wn;  bn.u[1] = *(const uint4*)(wn + 8);
                accR = __builtin_amdgcn_wmma_f32_16x16x32_bf16(false, afr[kc].v, false, br.v, (short)0, accR, false, false);
                accZ = __builtin_amdgcn_wmma_f32_16x16x32_bf16(false, afr[kc].v, false, bz.v, (short)0, accZ, false, false);
                accN = __builtin_amdgcn_wmma_f32_16x16x32_bf16(false, afr[kc].v, false, bn.v, (short)0, accN, false, false);
            }

            // per-column parameters (uniform over the 8 rows)
            const float wihR = w_ih[c],       bihR = b_ih[c],       bhhR = b_hh[c];
            const float wihZ = w_ih[256 + c], bihZ = b_ih[256 + c], bhhZ = b_hh[256 + c];
            const float wihN = w_ih[512 + c], bihN = b_ih[512 + c], bhhN = b_hh[512 + c];
            const float wo   = w_out[c];

            #pragma unroll
            for (int v = 0; v < 8; ++v) {
                const int mr = v + 8 * hi;         // row in tile (C layout)
                const int li = mr * Hdim + c;
                const float x = xq[v];
                float r = 1.0f / (1.0f + __expf(-(accR[v] + bhhR + x * wihR + bihR)));
                float z = 1.0f / (1.0f + __expf(-(accZ[v] + bhhZ + x * wihZ + bihZ)));
                float n = tanhf(x * wihN + bihN + r * (accN[v] + bhhN));
                float hin = h_w[li] + e_w[li];
                float hn  = (1.0f - z) * n + z * hin;
                h_w[li] = hn;
                op[v] += hn * wo;
            }
        }

        // ---- butterfly-reduce o across the 16 column lanes ----
        // after this, EVERY lane in a 16-lane half holds the full sums for
        // rows v + 8*hi -> feedback stays in registers, no LDS round-trip
        #pragma unroll
        for (int v = 0; v < 8; ++v) {
            float t = op[v];
            t += __shfl_xor(t, 1, 16);
            t += __shfl_xor(t, 2, 16);
            t += __shfl_xor(t, 4, 16);
            t += __shfl_xor(t, 8, 16);
            op[v] = t;
        }

        #pragma unroll
        for (int v = 0; v < 8; ++v) {
            float o = op[v] + bout;
            float q = (o > 0.0f) ? 1.0f : -1.0f;
            xq[v] = q;                              // next-step feedback (all lanes)
            if (ln16 == 0) {                        // lanes 0 and 16 store their rows
                const int g = row0 + v + 8 * hi;
                out[(size_t)g * NOUT + s]               = o;
                out[(size_t)BLN + (size_t)g * NOUT + s] = q;
            }
        }
    }
}

extern "C" void kernel_launch(void* const* d_in, const int* in_sizes, int n_in,
                              void* d_out, int out_size, void* d_ws, size_t ws_size,
                              hipStream_t stream) {
    const float* E     = (const float*)d_in[0];
    const float* w_ih  = (const float*)d_in[1];
    const float* w_hh  = (const float*)d_in[2];
    const float* b_ih  = (const float*)d_in[3];
    const float* b_hh  = (const float*)d_in[4];
    const float* w_out = (const float*)d_in[5];
    const float* b_out = (const float*)d_in[6];
    float* out = (float*)d_out;
    unsigned short* Wbf = (unsigned short*)d_ws;

    // convert w_hh to bf16 once per call (deterministic)
    convert_whh_bf16<<<(NG * Hdim + 255) / 256, 256, 0, stream>>>(w_hh, Wbf, NG * Hdim);

    const size_t shmem = (size_t)ROWS_PER_WG * Hdim * 4   // h  f32   (64 KB)
                       + (size_t)ROWS_PER_WG * Hdim * 4   // e  f32   (64 KB)
                       + (size_t)ROWS_PER_WG * Hdim * 2;  // A  bf16  (32 KB) => 160 KB, 2 WG/WGP
    (void)hipFuncSetAttribute((const void*)gru_wmma_kernel,
                              hipFuncAttributeMaxDynamicSharedMemorySize, (int)shmem);

    gru_wmma_kernel<<<BLrows / ROWS_PER_WG, WAVES_PER_WG * 32, shmem, stream>>>(
        E, w_ih, b_ih, b_hh, w_out, b_out, Wbf, out);
}
